// CgpHmmCell_43791486550070
// MI455X (gfx1250) — matrix operand order repaired and checked by reading, not compiled, not run on hardware
//
#include <hip/hip_runtime.h>
#include <hip/hip_bf16.h>

#define BATCH 128
#define T_LEN 8192
#define NS    128      // states
#define NEMIT 64       // emission symbols
#define AH_STRIDE 136  // padded f16 row stride (16B aligned, conflict-free b128 reads)
#define AHSZ (16 * AH_STRIDE)
#define BE_STRIDE 132  // padded f32 row stride for emission matrix in LDS
#define CSCALE  1024.0f
#define RCSCALE (1.0f / 1024.0f)
#define LOGC    6.931471805599453f   // log(1024)

typedef _Float16 v16h __attribute__((ext_vector_type(16)));
typedef _Float16 v8h  __attribute__((ext_vector_type(8)));
typedef float    v8f  __attribute__((ext_vector_type(8)));

// Sum across each aligned 16-lane row via DPP16 VALU ops (epilogue only).
__device__ __forceinline__ float dpp_sum16(float x) {
  x += __int_as_float(__builtin_amdgcn_update_dpp(
         0, __float_as_int(x), 0xB1 /*quad_perm(1,0,3,2)*/, 0xF, 0xF, true));
  x += __int_as_float(__builtin_amdgcn_update_dpp(
         0, __float_as_int(x), 0x4E /*quad_perm(2,3,0,1)*/, 0xF, 0xF, true));
  x += __int_as_float(__builtin_amdgcn_update_dpp(
         0, __float_as_int(x), 0x141 /*row_half_mirror*/, 0xF, 0xF, true));
  x += __int_as_float(__builtin_amdgcn_update_dpp(
         0, __float_as_int(x), 0x140 /*row_mirror*/, 0xF, 0xF, true));
  return x;
}

// ---------------------------------------------------------------------------
// Pass 1: collapse one-hot [B][T][64] f32 -> transposed index array idxT[T][B]
// ---------------------------------------------------------------------------
__global__ __launch_bounds__(256) void hmm_extract_idx(
    const float* __restrict__ x, int* __restrict__ idxT) {
  long gid = (long)blockIdx.x * blockDim.x + threadIdx.x;
  if (gid >= (long)BATCH * T_LEN) return;
  int b = (int)(gid / T_LEN);
  int t = (int)(gid % T_LEN);
  const float* p = x + gid * NEMIT;   // 256B-aligned
  int idx = 0;
  #pragma unroll
  for (int e = 0; e < NEMIT; e += 4) {
    float4 v = *(const float4*)(p + e);
    if (v.x != 0.f) idx = e;
    if (v.y != 0.f) idx = e + 1;
    if (v.z != 0.f) idx = e + 2;
    if (v.w != 0.f) idx = e + 3;
  }
  idxT[(long)t * BATCH + b] = idx;
}

// ---------------------------------------------------------------------------
// Pass 2: sequential scan, ONE barrier per step. Row sums W_{t-1} come from a
// second WMMA chain against a constant all-ones B operand (rowsum-as-matmul):
// no cross-wave exchange, no DPP, no branches in the hot loop.
// 8x v_wmma_f32_16x16x32_f16 per step per wave.
// ---------------------------------------------------------------------------
__global__ __launch_bounds__(256) void hmm_scan(
    const int*   __restrict__ idxT,  // [T][B]
    const float* __restrict__ Iv,    // [128]
    const float* __restrict__ Ag,    // [128][128] transition
    const float* __restrict__ Bg,    // [64][128]  emission
    float*       __restrict__ out)   // alpha[128][128] ++ loglik[128]
{
  __shared__ __align__(16) _Float16 sbuf[2 * AHSZ];         // S_t = c*(E (.) (beta@A)), f16
  __shared__ __align__(16) float Bem[NEMIT * BE_STRIDE];    // c * emission (~33 KB)
  __shared__ __align__(16) float If[NS];
  __shared__ __align__(16) float llk[16];                   // running sum log W per row
  __shared__ __align__(16) int   idxb[2 * 16];              // [2][row] pre-scaled offsets

  const int tid  = threadIdx.x;
  const int lane = tid & 31;
  const int w    = tid >> 5;          // wave id == N tile id (0..7)
  const int hi   = (lane >> 4) & 1;   // lane half
  const int l16  = lane & 15;
  const int b0   = blockIdx.x * 16;
  const int ncol = w * 16 + l16;

  // stage CSCALE-pre-multiplied emission matrix + I into LDS
  for (int i = tid; i < NEMIT * NS; i += 256)
    Bem[(i / NS) * BE_STRIDE + (i % NS)] = Bg[i] * CSCALE;
  for (int i = tid; i < NS; i += 256) If[i] = Iv[i];

  // HMM transition A in registers, WMMA B-operand layout (constant all steps).
  v16h bmat[4];
  {
    const int kb = hi * 8;
    #pragma unroll
    for (int kc = 0; kc < 4; ++kc) {
      v16h b;
      #pragma unroll
      for (int i = 0; i < 16; ++i)
        b[i] = (_Float16)Ag[(kc * 32 + kb + i + (i >= 8 ? 8 : 0)) * NS + ncol];
      bmat[kc] = b;
    }
  }
  // constant all-ones B operand: rowsum-as-matmul
  v16h ones;
  #pragma unroll
  for (int i = 0; i < 16; ++i) ones[i] = (_Float16)1.0f;

  int idx_reg = 0;
  if (tid < 16) {
    idxb[tid] = idxT[b0 + tid] * BE_STRIDE;   // slot 0: t = 0 (pre-scaled)
    idx_reg   = idxT[BATCH + b0 + tid];       // t = 1
    llk[tid]  = 0.f;
  }
  __syncthreads();

  // ---- t == 0: S_0 = (c*Bem_raw[idx0]) (.) I stored f16; W_0 comes from the
  //      ones-WMMA at t == 1, so no reduction is needed here. ----
  {
    int m = tid >> 4, j = tid & 15;
    int id0 = idxb[m];                        // already * BE_STRIDE
    #pragma unroll
    for (int c = 0; c < 8; ++c) {
      int col = j * 8 + c;
      float sc = Bem[id0 + col] * If[col];    // CSCALE folded into Bem
      sbuf[m * AH_STRIDE + col] = (_Float16)sc;
    }
  }
  // no barrier needed: first loop iteration's barrier orders these writes.

  int off_r = 0;    // sbuf offset holding S_{t-1}
  for (int t = 1; t < T_LEN; ++t) {
    // ---- publish pre-scaled indices for step t; prefetch t+1 ----
    if (tid < 16) {
      idxb[(t & 1) * 16 + tid] = idx_reg * BE_STRIDE;
      int tn = (t + 1 < T_LEN) ? t + 1 : t;
      idx_reg = idxT[(long)tn * BATCH + b0 + tid];
    }
    __syncthreads();                          // idx_cur + S_{t-1} visible

    // ---- A-operand loads first (feeds WMMA ASAP) ----
    const _Float16* sold = &sbuf[off_r];
    v8h alo[4], ahi[4];
    #pragma unroll
    for (int kc = 0; kc < 4; ++kc) {
      const _Float16* ap = sold + l16 * AH_STRIDE + kc * 32 + hi * 8;
      alo[kc] = *(const v8h*)ap;
      ahi[kc] = *(const v8h*)(ap + 16);
    }

    // ---- emission gathers (independent of WMMA results); c pre-folded ----
    const int* ic = &idxb[(t & 1) * 16];
    int4 ia = *(const int4*)&ic[hi * 8];
    int4 ib = *(const int4*)&ic[hi * 8 + 4];
    int id[8] = {ia.x, ia.y, ia.z, ia.w, ib.x, ib.y, ib.z, ib.w};
    float ev[8];
    #pragma unroll
    for (int r = 0; r < 8; ++r)
      ev[r] = Bem[id[r] + ncol];

    // ---- dual WMMA chains: acc = S@A ; acc2 = S@ones = row sums W_{t-1} ----
    v8f acc = {}, acc2 = {};
    #pragma unroll
    for (int kc = 0; kc < 4; ++kc) {
      v16h a = __builtin_shufflevector(alo[kc], ahi[kc],
                 0, 1, 2, 3, 4, 5, 6, 7, 8, 9, 10, 11, 12, 13, 14, 15);
      acc2 = __builtin_amdgcn_wmma_f32_16x16x32_f16(
          false, a, false, ones, (short)0, acc2, false, false);
      acc  = __builtin_amdgcn_wmma_f32_16x16x32_f16(
          false, a, false, bmat[kc], (short)0, acc, false, false);
    }

    // ---- loglik: every lane holds W for its 8 rows; wave 0 row-lanes log it ----
    if (w == 0 && l16 < 8) {
      float Wm = acc2[0];
      #pragma unroll
      for (int r = 1; r < 8; ++r) Wm = (l16 == r) ? acc2[r] : Wm;
      llk[hi * 8 + l16] += __logf(Wm) - LOGC;   // LDS RMW, wave-ordered
    }

    // ---- S_t = acc * (c*e) / W ; store f16 into other buffer ----
    _Float16* snew = &sbuf[off_r ^ AHSZ];
    #pragma unroll
    for (int r = 0; r < 8; ++r) {
      float sc = acc[r] * (ev[r] * __builtin_amdgcn_rcpf(acc2[r]));
      snew[(hi * 8 + r) * AH_STRIDE + ncol] = (_Float16)sc;
    }

    off_r ^= AHSZ;
  }
  __syncthreads();

  // ---- epilogue: final UNSCALED alpha = S_{T-1}/c, plus last loglik term
  //      log(sum(alpha_f)) fused into the same pass ----
  {
    int m = tid >> 4, j = tid & 15;
    const _Float16* sp = &sbuf[off_r + m * AH_STRIDE + j * 8];
    float s = 0.f;
    #pragma unroll
    for (int c = 0; c < 8; ++c) {
      float v = (float)sp[c] * RCSCALE;
      out[(long)(b0 + m) * NS + j * 8 + c] = v;
      s += v;
    }
    float W = dpp_sum16(s);                  // full row sum of final alpha
    if (j == 0) out[BATCH * NS + b0 + m] = llk[m] + __logf(W);
  }
}

// ---------------------------------------------------------------------------
extern "C" void kernel_launch(void* const* d_in, const int* in_sizes, int n_in,
                              void* d_out, int out_size, void* d_ws, size_t ws_size,
                              hipStream_t stream) {
  const float* x  = (const float*)d_in[0];  // inputs [B,T,EMIT] one-hot f32
  const float* Iv = (const float*)d_in[1];  // [128]
  const float* Ag = (const float*)d_in[2];  // [128,128]
  const float* Bg = (const float*)d_in[3];  // [64,128]
  int* idxT = (int*)d_ws;                   // [T][B] scratch (4 MB)

  long n = (long)BATCH * T_LEN;
  hmm_extract_idx<<<(int)((n + 255) / 256), 256, 0, stream>>>(x, idxT);
  hmm_scan<<<BATCH / 16, 256, 0, stream>>>(idxT, Iv, Ag, Bg, (float*)d_out);
}
